// AttnProcessor_LoRA_Capture_86517821216225
// MI455X (gfx1250) — matrix operand order repaired and checked by reading, not compiled, not run on hardware
//
#include <hip/hip_runtime.h>
#include <hip/hip_bf16.h>

#define BB 8
#define LQN 4096
#define SS 77
#define DD 1024
#define HH 16
#define HD 64
#define RR 192
#define NSUBJ 32
static const float LORA_SCALE_C = 16.0f / 192.0f;

typedef __attribute__((ext_vector_type(16))) __bf16 v16bf;
typedef __attribute__((ext_vector_type(8)))  float  v8f;
typedef __attribute__((ext_vector_type(4)))  unsigned int v4u;
typedef __attribute__((ext_vector_type(8)))  int v8i;
typedef __attribute__((ext_vector_type(4)))  int v4i;

#if defined(__HIP_DEVICE_COMPILE__) && __has_builtin(__builtin_amdgcn_tensor_load_to_lds) && __has_builtin(__builtin_amdgcn_s_wait_tensorcnt)
#define HAVE_TDM 1
#else
#define HAVE_TDM 0
#endif

// ---------- bf16 helpers (native HW converts) ----------
__device__ inline unsigned short bfraw(__bf16 b) { return __builtin_bit_cast(unsigned short, b); }
__device__ inline float bf2f(unsigned short h) {
  unsigned int u = ((unsigned int)h) << 16;
  return __builtin_bit_cast(float, u);
}
__device__ inline __bf16 tobf(float x) { return (__bf16)x; }
__device__ inline __bf16 tobf(unsigned short x) { return __builtin_bit_cast(__bf16, x); }

// ---------- WMMA fragment loaders (gfx1250 16x16x32 bf16 layouts, ISA 7.12.2) ----------
// GUARD=false: bounds statically known satisfied -> straight-line b128 loads, no exec masking.
// A (16x32, MxK): lanes 0-15 -> M=lane, K = kb+{0..7,16..23}; lanes 16-31 -> M=lane-16, K = kb+{8..15,24..31}
template <bool GUARD, typename T>
__device__ inline v16bf load_a_frag(const T* A, int lda, int mbase, int Mlim, int kb) {
  int lane = threadIdx.x & 31;
  int m = mbase + (lane & 15);
  int ko = kb + ((lane >> 4) << 3);
  v16bf f;
  if (!GUARD || m < Mlim) {
    const T* p = A + (long)m * lda + ko;
#pragma unroll
    for (int j = 0; j < 8; ++j) { f[j] = tobf(p[j]); f[j + 8] = tobf(p[16 + j]); }
  } else {
#pragma unroll
    for (int j = 0; j < 16; ++j) f[j] = (__bf16)0.0f;
  }
  return f;
}

// B (32x16, KxN): lanes 0-15 -> N=lane, K = kb+{0..7,16..23}; lanes 16-31 -> N=lane-16, K = kb+{8..15,24..31}
template <bool GUARD, typename T>
__device__ inline v16bf load_b_frag(const T* Bm, int ldb, int nbase, int Nlim, int kb) {
  int lane = threadIdx.x & 31;
  int n = nbase + (lane & 15);
  int ko = kb + ((lane >> 4) << 3);
  v16bf f;
  if (!GUARD || n < Nlim) {
    const T* p = Bm + (long)n * ldb + ko;
#pragma unroll
    for (int j = 0; j < 8; ++j) { f[j] = tobf(p[j]); f[j + 8] = tobf(p[16 + j]); }
  } else {
#pragma unroll
    for (int j = 0; j < 16; ++j) f[j] = (__bf16)0.0f;
  }
  return f;
}

__device__ inline v8f wmma_bf16(v16bf a, v16bf b, v8f c) {
  return __builtin_amdgcn_wmma_f32_16x16x32_bf16(false, a, false, b, (short)0, c, false, false);
}

// ---------- f32 -> bf16 bulk conversion (bandwidth bound, vectorized) ----------
__global__ __launch_bounds__(256) void cvt_f32_to_bf16(const float* __restrict__ in,
                                                       unsigned short* __restrict__ out, long n) {
  long i = ((long)blockIdx.x * 256 + threadIdx.x) * 4;
  if (i + 3 < n) {
    float4 f = *(const float4*)(in + i);
    unsigned int lo = (unsigned int)bfraw((__bf16)f.x) | ((unsigned int)bfraw((__bf16)f.y) << 16);
    unsigned int hi = (unsigned int)bfraw((__bf16)f.z) | ((unsigned int)bfraw((__bf16)f.w) << 16);
    *(uint2*)(out + i) = make_uint2(lo, hi);
  } else {
    for (long j = i; j < n; ++j) out[j] = bfraw((__bf16)in[j]);
  }
}

// ---------- GEMM: out[m,n] = alpha * sum_k A[m,k]*B[n,k], bf16 in, bf16 out ----------
// block = 4 waves; wave computes 16 rows x 64 cols (A-frag reused 4x). MG: m-bounds guard.
template <bool MG>
__global__ __launch_bounds__(128) void gemm_abT_bf16(const unsigned short* __restrict__ A, int lda,
                                                     const unsigned short* __restrict__ B, int ldb,
                                                     unsigned short* __restrict__ out, int ldo,
                                                     int M, int N, int K, float alpha) {
  int lane = threadIdx.x & 31, wave = threadIdx.x >> 5;
  int mbase = blockIdx.x * 64 + wave * 16;
  int nb0 = blockIdx.y * 64;
  v8f acc[4] = {{}, {}, {}, {}};
  for (int kb = 0; kb < K; kb += 32) {
    v16bf a = load_a_frag<MG>(A, lda, mbase, M, kb);
    if (kb + 32 < K) __builtin_prefetch(B + (long)(nb0 + (lane & 15)) * ldb + kb + 32, 0, 1);
#pragma unroll
    for (int t = 0; t < 4; ++t) {
      v16bf b = load_b_frag<false>(B, ldb, nb0 + t * 16, N, kb);
      acc[t] = wmma_bf16(a, b, acc[t]);
    }
  }
  int hi = lane >> 4, nl = lane & 15;
#pragma unroll
  for (int t = 0; t < 4; ++t) {
    int n = nb0 + t * 16 + nl;
#pragma unroll
    for (int r = 0; r < 8; ++r) {
      int m = mbase + r + hi * 8;
      if (!MG || m < M) out[(long)m * ldo + n] = bfraw((__bf16)(alpha * acc[t][r]));
    }
  }
}

// ---------- Dual GEMM: out[m,n] = sum_k A1[m,k]*B1[n,k] + sum_r A2[m,r]*B2[n,r] (+bias) ----------
template <bool OUTF32, bool MG>
__global__ __launch_bounds__(128) void gemm_dual(const unsigned short* __restrict__ A1, int lda1,
                                                 const unsigned short* __restrict__ A2, int lda2,
                                                 const unsigned short* __restrict__ B1, int ldb1,
                                                 const unsigned short* __restrict__ B2, int ldb2,
                                                 const float* __restrict__ bias,
                                                 float* __restrict__ outf,
                                                 unsigned short* __restrict__ outh, int ldo,
                                                 int M, int N, int K1, int K2) {
  int lane = threadIdx.x & 31, wave = threadIdx.x >> 5;
  int mbase = blockIdx.x * 64 + wave * 16;
  int nb0 = blockIdx.y * 64;
  v8f acc[4] = {{}, {}, {}, {}};
  for (int kb = 0; kb < K1; kb += 32) {
    v16bf a = load_a_frag<MG>(A1, lda1, mbase, M, kb);
    if (kb + 32 < K1) __builtin_prefetch(B1 + (long)(nb0 + (lane & 15)) * ldb1 + kb + 32, 0, 1);
#pragma unroll
    for (int t = 0; t < 4; ++t) {
      v16bf b = load_b_frag<false>(B1, ldb1, nb0 + t * 16, N, kb);
      acc[t] = wmma_bf16(a, b, acc[t]);
    }
  }
  for (int kb = 0; kb < K2; kb += 32) {
    v16bf a = load_a_frag<MG>(A2, lda2, mbase, M, kb);
#pragma unroll
    for (int t = 0; t < 4; ++t) {
      v16bf b = load_b_frag<false>(B2, ldb2, nb0 + t * 16, N, kb);
      acc[t] = wmma_bf16(a, b, acc[t]);
    }
  }
  int hi = lane >> 4, nl = lane & 15;
#pragma unroll
  for (int t = 0; t < 4; ++t) {
    int n = nb0 + t * 16 + nl;
    float bs = bias ? bias[n] : 0.0f;
#pragma unroll
    for (int r = 0; r < 8; ++r) {
      int m = mbase + r + hi * 8;
      if (!MG || m < M) {
        float v = acc[t][r] + bs;
        if constexpr (OUTF32) outf[(long)m * ldo + n] = v;
        else outh[(long)m * ldo + n] = bfraw((__bf16)v);
      }
    }
  }
}

// ---------- qsum[b, c] = sum_l q_bf16[b, l, c] (q already has 1/sqrt(HD) folded in) ----------
__global__ __launch_bounds__(256) void qsum_kernel(const unsigned short* __restrict__ q,
                                                   float* __restrict__ qsum) {
  int b = blockIdx.x;
  int c = blockIdx.y * 256 + threadIdx.x;
  const unsigned short* p = q + (long)b * LQN * DD + c;
  float s = 0.f;
  for (int l = 0; l < LQN; ++l) s += bf2f(p[(long)l * DD]);
  qsum[b * DD + c] = s;
}

// ---------- per-(b,h,n) column mean of attn scores for subject tokens ----------
__global__ void subj_kernel(const float* __restrict__ qsum, const unsigned short* __restrict__ k,
                            const int* __restrict__ sb, const int* __restrict__ sn,
                            float* __restrict__ means, int* __restrict__ flags) {
  int t = threadIdx.x;
  if (t >= NSUBJ * HH) return;
  int i = t >> 4, h = t & 15;
  int b = sb[i], n = sn[i];
  if (b < 0 || b >= BB || n < 0 || n >= SS) return;
  const float* qs = qsum + b * DD + h * HD;
  const unsigned short* kp = k + ((long)b * SS + n) * DD + h * HD;
  float d = 0.f;
#pragma unroll 8
  for (int j = 0; j < HD; ++j) d += qs[j] * bf2f(kp[j]);
  means[(b * HH + h) * SS + n] = d / (float)LQN;
  if (h == 0) flags[b * SS + n] = 1;
}

// ---------- v transpose: v_t[b][h][d][96] (bf16, zero-padded s in [77,96)) ----------
__global__ __launch_bounds__(256) void transpose_v(const unsigned short* __restrict__ v,
                                                   unsigned short* __restrict__ v_t) {
  const int b = blockIdx.x, h = blockIdx.y;
  unsigned short* dst = v_t + (long)(b * HH + h) * 64 * 96;
  for (int i = threadIdx.x; i < 64 * 96; i += 256) {
    int d = i / 96, s = i - 96 * d;
    unsigned short val = 0;
    if (s < SS) val = v[((long)b * SS + s) * DD + h * HD + d];
    dst[i] = val;
  }
}

// ---------- fused attention: block = 8 waves, each wave owns a 16-row l-tile of one (b,h) ----------
__global__ __launch_bounds__(256) void attn_kernel(const unsigned short* __restrict__ q,
                                                   const unsigned short* __restrict__ k,
                                                   const unsigned short* __restrict__ v_t,
                                                   const float* __restrict__ means,
                                                   const int* __restrict__ flags,
                                                   const float* __restrict__ caf_p,
                                                   unsigned short* __restrict__ ctx) {
  const int b = blockIdx.z, h = blockIdx.y;
  const int l0 = blockIdx.x * 128;
  __shared__ unsigned short v_lds[64 * 96];  // [d][s], bf16, s padded to 96 (pad pre-zeroed in v_t)
  __shared__ float sc[8][16 * 96];           // per-wave score/P tile, rows x 96
  __shared__ float rinv[8][16];
  int tid = threadIdx.x, lane = tid & 31, wave = tid >> 5;

  const unsigned short* vtile = v_t + (long)(b * HH + h) * 64 * 96;
#if HAVE_TDM
  if (wave == 0) {
    // Tensor Data Mover: one 12KB contiguous tile -> LDS (D# per ISA ch.8)
    unsigned long long ga = (unsigned long long)(const void*)vtile;
    unsigned int ldsoff = (unsigned int)(unsigned long long)(const void*)&v_lds[0];
    v4u g0;
    g0.x = 1u;                                            // count = 1 valid descriptor
    g0.y = ldsoff;                                        // lds_addr (bytes)
    g0.z = (unsigned int)(ga & 0xffffffffu);              // global_addr[31:0]
    g0.w = (unsigned int)((ga >> 32) & 0x01ffffffu) | 0x80000000u;  // addr[56:32] | type=2
    v8i g1;
    g1[0] = 0x00030000;   // workgroup_mask=0, data_size=3 (8B units)
    g1[1] = 0x06000000;   // tensor_dim0[15:0] = 1536 (64*96*2B / 8B)
    g1[2] = 0x00010000;   // tensor_dim1 = 1
    g1[3] = 0x06000000;   // tile_dim0 = 1536
    g1[4] = 0x00000001;   // tile_dim1 = 1
    g1[5] = 1536;         // tensor_dim0_stride
    g1[6] = 0; g1[7] = 0;
    v4i z4 = {0, 0, 0, 0};
#if defined(__clang_major__) && (__clang_major__ >= 23)
    v8i z8 = {0, 0, 0, 0, 0, 0, 0, 0};
    __builtin_amdgcn_tensor_load_to_lds(g0, g1, z4, z4, z8, 0);
#else
    __builtin_amdgcn_tensor_load_to_lds(g0, g1, z4, z4, 0);
#endif
    __builtin_amdgcn_s_wait_tensorcnt(0);
  }
#else
  for (int i = tid; i < 64 * 96; i += 256) v_lds[i] = vtile[i];
#endif
  __syncthreads();

  float caf = caf_p[0];
  int mrow = l0 + wave * 16;
  const unsigned short* qb = q + (long)b * LQN * DD + h * HD;
  const unsigned short* kbp = k + (long)b * SS * DD + h * HD;
  float* scw = &sc[wave][0];

  v16bf qa0 = load_a_frag<false>(qb, DD, mrow, LQN, 0);
  v16bf qa1 = load_a_frag<false>(qb, DD, mrow, LQN, 32);

  // zero pad columns 80..95 of the P tile
  for (int i = lane; i < 256; i += 32) { int r = i >> 4, c = 80 + (i & 15); scw[r * 96 + c] = 0.f; }

  int hi = lane >> 4, nl = lane & 15;
  for (int st = 0; st < 5; ++st) {
    v16bf kb0 = load_b_frag<true>(kbp, DD, st * 16, SS, 0);
    v16bf kb1 = load_b_frag<true>(kbp, DD, st * 16, SS, 32);
    v8f a = {};
    a = wmma_bf16(qa0, kb0, a);
    a = wmma_bf16(qa1, kb1, a);
    int n = st * 16 + nl;
    float mn = 0.f; int fl = 0;
    if (n < SS) { fl = flags[b * SS + n]; if (fl) mn = means[(b * HH + h) * SS + n]; }
#pragma unroll
    for (int r = 0; r < 8; ++r) {
      float x = a[r];
      if (n >= SS) x = -1e30f;
      else if (fl) x = (x - mn) * caf;
      scw[(r + hi * 8) * 96 + n] = x;
    }
  }

  // in-wave softmax over 77 columns; unnormalized exp left in LDS, 1/sum in rinv
  {
    int row = lane & 15, half = lane >> 4;
    float mx = -1e30f;
    for (int c = half * 40; c < half * 40 + 40; ++c) mx = fmaxf(mx, scw[row * 96 + c]);
    mx = fmaxf(mx, __shfl_xor(mx, 16));
    float sum = 0.f;
    for (int c = half * 40; c < half * 40 + 40; ++c) {
      float e = __expf(scw[row * 96 + c] - mx);
      scw[row * 96 + c] = e;
      sum += e;
    }
    sum += __shfl_xor(sum, 16);
    if (half == 0) rinv[wave][row] = 1.0f / sum;
  }

  // P x V via WMMA (K padded to 96; pad region exact zeros on both sides)
  v16bf pa0 = load_a_frag<false>(scw, 96, 0, 16, 0);
  v16bf pa1 = load_a_frag<false>(scw, 96, 0, 16, 32);
  v16bf pa2 = load_a_frag<false>(scw, 96, 0, 16, 64);

  unsigned short* ctxb = ctx + (long)b * LQN * DD + h * HD;
#pragma unroll
  for (int nt = 0; nt < 4; ++nt) {
    v16bf vb0 = load_b_frag<false>(v_lds, 96, nt * 16, 64, 0);
    v16bf vb1 = load_b_frag<false>(v_lds, 96, nt * 16, 64, 32);
    v16bf vb2 = load_b_frag<false>(v_lds, 96, nt * 16, 64, 64);
    v8f o = {};
    o = wmma_bf16(pa0, vb0, o);
    o = wmma_bf16(pa1, vb1, o);
    o = wmma_bf16(pa2, vb2, o);
#pragma unroll
    for (int r = 0; r < 8; ++r) {
      int m = r + hi * 8;
      float val = o[r] * rinv[wave][m];
      ctxb[(long)(mrow + m) * DD + nt * 16 + nl] = bfraw((__bf16)val);
    }
  }
}

// ---------- host launch ----------
extern "C" void kernel_launch(void* const* d_in, const int* in_sizes, int n_in,
                              void* d_out, int out_size, void* d_ws, size_t ws_size,
                              hipStream_t stream) {
  (void)in_sizes; (void)n_in; (void)out_size; (void)ws_size;
  const float* hs  = (const float*)d_in[0];
  const float* ehs = (const float*)d_in[1];
  const float* Wf[4]  = {(const float*)d_in[2], (const float*)d_in[3], (const float*)d_in[4], (const float*)d_in[5]};
  const float* bo  = (const float*)d_in[6];
  const float* Lf[6]  = {(const float*)d_in[7], (const float*)d_in[8], (const float*)d_in[9],
                         (const float*)d_in[10], (const float*)d_in[11], (const float*)d_in[12]};
  const float* caf = (const float*)d_in[13];
  const int* subj_b = (const int*)d_in[14];
  const int* subj_n = (const int*)d_in[15];
  float* out = (float*)d_out;

  char* ws = (char*)d_ws;
  size_t off = 0;
  auto alloc = [&](size_t bytes) -> void* {
    void* p = ws + off;
    off = (off + bytes + 255) & ~(size_t)255;
    return p;
  };
  typedef unsigned short u16;
  u16* qbuf  = (u16*)alloc((size_t)BB * LQN * DD * 2);
  u16* ctxb  = (u16*)alloc((size_t)BB * LQN * DD * 2);
  u16* hs_bf = (u16*)alloc((size_t)BB * LQN * DD * 2);
  u16* ehs_bf= (u16*)alloc((size_t)BB * SS * DD * 2);
  u16* kbuf  = (u16*)alloc((size_t)BB * SS * DD * 2);
  u16* vbuf  = (u16*)alloc((size_t)BB * SS * DD * 2);
  u16* t_k   = (u16*)alloc((size_t)BB * SS * RR * 2);
  u16* t_v   = (u16*)alloc((size_t)BB * SS * RR * 2);
  u16* t_o   = (u16*)alloc((size_t)BB * LQN * RR * 2);
  u16* v_t   = (u16*)alloc((size_t)BB * HH * 64 * 96 * 2);
  u16* Wb[4]; for (int i = 0; i < 4; ++i) Wb[i] = (u16*)alloc((size_t)DD * DD * 2);
  u16* Lb[6]; for (int i = 0; i < 6; ++i) Lb[i] = (u16*)alloc((size_t)DD * RR * 2);
  float* qsum  = (float*)alloc((size_t)BB * DD * 4);
  float* means = (float*)alloc((size_t)BB * HH * SS * 4);
  int*   flags = (int*)alloc((size_t)BB * SS * 4);

  // bulk f32 -> bf16 conversions (one pass; GEMMs then stream pure bf16)
  auto cvt = [&](const float* src, u16* dst, long n) {
    long blocks = (n + 1023) / 1024;
    cvt_f32_to_bf16<<<dim3((unsigned)blocks), 256, 0, stream>>>(src, dst, n);
  };
  cvt(hs, hs_bf, (long)BB * LQN * DD);
  cvt(ehs, ehs_bf, (long)BB * SS * DD);
  for (int i = 0; i < 4; ++i) cvt(Wf[i], Wb[i], (long)DD * DD);
  for (int i = 0; i < 6; ++i) cvt(Lf[i], Lb[i], (long)DD * RR);
  u16 *Wq_b = Wb[0], *Wk_b = Wb[1], *Wv_b = Wb[2], *Wo_b = Wb[3];
  u16 *Ak_b = Lb[0], *Bk_b = Lb[1], *Av_b = Lb[2], *Bv_b = Lb[3], *Ao_b = Lb[4], *Bo_b = Lb[5];

  const int Mkv = BB * SS;           // 616 (needs m-guard)
  const int Mq  = BB * LQN;          // 32768 (full tiles, unguarded)
  const int gKV = (Mkv + 63) / 64;   // 10
  const int gQ  = Mq / 64;           // 512

  // LoRA rank-R intermediates (pre-scaled by alpha/R)
  gemm_abT_bf16<true><<<dim3(gKV, RR / 64), 128, 0, stream>>>(ehs_bf, DD, Ak_b, DD, t_k, RR, Mkv, RR, DD, LORA_SCALE_C);
  gemm_abT_bf16<true><<<dim3(gKV, RR / 64), 128, 0, stream>>>(ehs_bf, DD, Av_b, DD, t_v, RR, Mkv, RR, DD, LORA_SCALE_C);
  // k = ehs*Wk^T + t_k*Bk^T ; v likewise
  gemm_dual<false, true><<<dim3(gKV, DD / 64), 128, 0, stream>>>(ehs_bf, DD, t_k, RR, Wk_b, DD, Bk_b, RR, nullptr, nullptr, kbuf, DD, Mkv, DD, DD, RR);
  gemm_dual<false, true><<<dim3(gKV, DD / 64), 128, 0, stream>>>(ehs_bf, DD, t_v, RR, Wv_b, DD, Bv_b, RR, nullptr, nullptr, vbuf, DD, Mkv, DD, DD, RR);
  // q = (1/sqrt(HD)) * hs*Wq^T  (attention scale folded in; no guards, full tiles)
  gemm_abT_bf16<false><<<dim3(gQ, DD / 64), 128, 0, stream>>>(hs_bf, DD, Wq_b, DD, qbuf, DD, Mq, DD, DD, 0.125f);
  // column-mean machinery for subject tokens
  qsum_kernel<<<dim3(BB, DD / 256), 256, 0, stream>>>(qbuf, qsum);
  hipMemsetAsync(flags, 0, (size_t)BB * SS * sizeof(int), stream);
  subj_kernel<<<1, 512, 0, stream>>>(qsum, kbuf, subj_b, subj_n, means, flags);
  // padded head-transposed v tiles for TDM staging
  transpose_v<<<dim3(BB, HH), 256, 0, stream>>>(vbuf, v_t);
  // fused attention
  attn_kernel<<<dim3(LQN / 128, HH, BB), 256, 0, stream>>>(qbuf, kbuf, v_t, means, flags, caf, ctxb);
  // output LoRA intermediate (pre-scaled; no guards)
  gemm_abT_bf16<false><<<dim3(gQ, RR / 64), 128, 0, stream>>>(ctxb, DD, Ao_b, DD, t_o, RR, Mq, RR, DD, LORA_SCALE_C);
  // out = ctx*Wo^T + bo + t_o*Bo^T  (f32 out; no guards)
  gemm_dual<true, false><<<dim3(gQ, DD / 64), 128, 0, stream>>>(ctxb, DD, t_o, RR, Wo_b, DD, Bo_b, RR, bo, out, nullptr, DD, Mq, DD, DD, RR);
}
